// FaceEmbedding_9440338117364
// MI455X (gfx1250) — compile-verified, matrix operand order checked by reading, not run on hardware
//
#include <hip/hip_runtime.h>

// ---------------------------------------------------------------------------
// FaceEmbedding for MI455X (gfx1250): bf16 WMMA GEMMs with f32 accumulation,
// async global->LDS staging (ASYNCcnt), 4-wave blocks with 64x64 per wave
// (16 WMMA tiles/wave -> 1:1 ds_load:wmma ratio).
// Grid is (col-blocks, row-blocks): the 3 col-blocks sharing one row-slice of
// x / t are dispatch-adjacent, so the slice is read from HBM once, L2 after.
//   prep:  W1T[384][704] = bf16(W1g^T) zero-padded; W2T[384][768] = bf16([W2g;W2e]^T)
//   t[:, 0:384]   = elu(x.reshape(N,700) @ W1g + b1g)        (GEMM1, K=704 pad)
//   t[:, 384:768] = elu(W1e[et] + area*W1e[6] + b1e)         (gather+axpy)
//   out           = mask ? t @ [W2g;W2e] + (b2g+b2e) : 0     (GEMM2, K=768)
// ---------------------------------------------------------------------------

#define N_NODES   100000
#define HID       384
#define KGRID     700
#define K1_PAD    704
#define K1_TILES  22
#define K2_DIM    768
#define K2_TILES  24
#define BM        128
#define BN        128
#define BK        32
#define LDST      40          // LDS row stride (elements): 80B, 16B-aligned, bank-conflict-free
#define NPAD      100096      // 782 * 128
#define ROW_BLKS  782
#define NT        128         // threads per GEMM block (4 waves)

typedef __bf16 bfx2  __attribute__((ext_vector_type(2)));
typedef __bf16 bfx4  __attribute__((ext_vector_type(4)));
typedef __bf16 bfx8  __attribute__((ext_vector_type(8)));
typedef __bf16 v16bf __attribute__((ext_vector_type(16)));
typedef float  v8f   __attribute__((ext_vector_type(8)));

static __device__ __forceinline__ __bf16 f2bf(float f) {
    unsigned u = __builtin_bit_cast(unsigned, f);
    unsigned r = (u + 0x7FFFu + ((u >> 16) & 1u)) >> 16;   // RNE
    unsigned short s = (unsigned short)r;
    return __builtin_bit_cast(__bf16, s);
}

// Packed f32x4 -> bf16x4 (hardware v_cvt_pk_bf16_f32 when available)
static __device__ __forceinline__ bfx4 cvt4(float a, float b, float c, float d) {
#if __has_builtin(__builtin_amdgcn_cvt_pk_bf16_f32)
    bfx2 lo = __builtin_amdgcn_cvt_pk_bf16_f32(a, b);
    bfx2 hi = __builtin_amdgcn_cvt_pk_bf16_f32(c, d);
    bfx4 r; r[0] = lo[0]; r[1] = lo[1]; r[2] = hi[0]; r[3] = hi[1];
    return r;
#else
    bfx4 r; r[0] = f2bf(a); r[1] = f2bf(b); r[2] = f2bf(c); r[3] = f2bf(d);
    return r;
#endif
}

static __device__ __forceinline__ float elu1(float v) {
    return v > 0.0f ? v : (__expf(v) - 1.0f);
}

// CDNA5 async copy: 16B global -> LDS per lane, tracked by ASYNCcnt.
static __device__ __forceinline__ void async_b128(const void* gsrc, void* ldst) {
    unsigned lds_off = (unsigned)(size_t)ldst;             // low 32 bits = LDS offset
    unsigned long long ga = (unsigned long long)(size_t)gsrc;
    asm volatile("global_load_async_to_lds_b128 %0, %1, off"
                 :: "v"(lds_off), "v"(ga) : "memory");
}
static __device__ __forceinline__ void wait_async0() {
    asm volatile("s_wait_asynccnt 0x0" ::: "memory");
}

// 16-elem bf16 fragment: K 0..7 at p, K 16..23 at p+16 (lane<16); K 8..15 /
// 24..31 for lane>=16 -- caller passes khalf-adjusted p.
static __device__ __forceinline__ v16bf load_frag(const __bf16* p) {
    bfx8 lo = *(const bfx8*)(p);
    bfx8 hi = *(const bfx8*)(p + 16);
    v16bf r;
#pragma unroll
    for (int i = 0; i < 8; ++i) { r[i] = lo[i]; r[8 + i] = hi[i]; }
    return r;
}

// ---------------------------------------------------------------------------
// Weight prep (tiny, once per launch): bf16 transposed + zero-padded weights.
// ---------------------------------------------------------------------------
__global__ __launch_bounds__(256) void face_prep_w1_kernel(
    const float* __restrict__ W1g, __bf16* __restrict__ W1T)
{
    int idx = blockIdx.x * 256 + threadIdx.x;            // over 384*704
    if (idx >= HID * K1_PAD) return;
    int n = idx / K1_PAD, k = idx % K1_PAD;
    float v = (k < KGRID) ? W1g[(long)k * HID + n] : 0.0f;
    W1T[idx] = f2bf(v);
}

__global__ __launch_bounds__(256) void face_prep_w2_kernel(
    const float* __restrict__ W2g, const float* __restrict__ W2e,
    __bf16* __restrict__ W2T)
{
    int idx = blockIdx.x * 256 + threadIdx.x;            // over 384*768
    if (idx >= HID * K2_DIM) return;
    int n = idx / K2_DIM, k = idx % K2_DIM;
    float v = (k < HID) ? W2g[(long)k * HID + n] : W2e[(long)(k - HID) * HID + n];
    W2T[idx] = f2bf(v);
}

// ---------------------------------------------------------------------------
// Entity branch (vectorized x4): t[n, 384+j..j+3]
// ---------------------------------------------------------------------------
__global__ __launch_bounds__(256) void face_entity_kernel(
    const int* __restrict__ et, const float* __restrict__ area,
    const float* __restrict__ W1e, const float* __restrict__ b1e,
    __bf16* __restrict__ t)
{
    long idx = (long)blockIdx.x * 256 + threadIdx.x;     // over N * 96
    if (idx >= (long)N_NODES * (HID / 4)) return;
    int n = (int)(idx / (HID / 4));
    int j = (int)(idx % (HID / 4)) * 4;
    int e = et[n];
    float ar = area[n];
    float4 w  = *(const float4*)(W1e + (long)e * HID + j);
    float4 wa = *(const float4*)(W1e + 6L * HID + j);
    float4 bb = *(const float4*)(b1e + j);
    float v0 = elu1(w.x + ar * wa.x + bb.x);
    float v1 = elu1(w.y + ar * wa.y + bb.y);
    float v2 = elu1(w.z + ar * wa.z + bb.z);
    float v3 = elu1(w.w + ar * wa.w + bb.w);
    *(bfx4*)(t + (long)n * K2_DIM + HID + j) = cvt4(v0, v1, v2, v3);
}

// ---------------------------------------------------------------------------
// GEMM1: t[:, :384] = bf16(elu(x @ W1g + b1g))
//   A: x f32 [N,700] -> cvt bf16 -> LDS (VGPR path, conversion needed);
//      K-tail (step 21, k=672..703) peeled out of the steady-state loop.
//   B: W1T bf16 [384][704] -> LDS via async copy (zero-padded tail).
// ---------------------------------------------------------------------------
__global__ __launch_bounds__(NT) void face_gemm1_kernel(
    const float* __restrict__ x, const __bf16* __restrict__ W1T,
    const float* __restrict__ b1g, __bf16* __restrict__ t)
{
    __shared__ alignas(16) __bf16 As[2][BM][LDST];   // (m,k)
    __shared__ alignas(16) __bf16 Bs[2][BN][LDST];   // (n,k)

    const int tid  = threadIdx.x;
    const int lane = tid & 31;
    const int wave = tid >> 5;
    const int wm   = wave & 1;        // 2 m-groups * 64 rows
    const int wn   = wave >> 1;       // 2 n-groups * 64 cols
    const int l16  = lane & 15;
    const int khalf = (lane >> 4) * 8;

    const int colBase = blockIdx.x * BN;   // 3 col-blocks: dispatch-adjacent
    const int rowBase = blockIdx.y * BM;   // -> x row-slice reused via L2

    v8f acc[4][4] = {};

    // staging thread mapping (128 threads)
    const int ar0 = tid >> 3;                  // 0..15 (A: row group)
    const int akg = (tid & 7) * 4;             // 0,4,..,28 (A: k offset)

    auto stageB = [&](int ks, int buf) {
#pragma unroll
        for (int p = 0; p < 4; ++p) {
            int idx = tid + NT * p;            // 0..511
            int n   = idx >> 2;                // 0..127
            int g   = (idx & 3) * 8;           // 0,8,16,24
            async_b128(W1T + (long)(colBase + n) * K1_PAD + ks * BK + g,
                       &Bs[buf][n][g]);
        }
    };
    auto stageAFull = [&](int ks, int buf) {   // ks in [0,21): fully in-bounds
        const int k0 = ks * BK + akg;
#pragma unroll
        for (int p = 0; p < 8; ++p) {
            int row = ar0 + 16 * p;
            int gr  = rowBase + row;
            if (gr >= N_NODES) gr = N_NODES - 1;
            float4 v = *(const float4*)(x + (long)gr * KGRID + k0);
            *(bfx4*)(&As[buf][row][akg]) = cvt4(v.x, v.y, v.z, v.w);
        }
    };
    auto stageATail = [&](int buf) {           // ks == 21: k = 672..703
        const int k0 = 21 * BK + akg;
        const bool inb = (akg != 28);          // k0=700..703 group is OOB
#pragma unroll
        for (int p = 0; p < 8; ++p) {
            int row = ar0 + 16 * p;
            int gr  = rowBase + row;
            if (gr >= N_NODES) gr = N_NODES - 1;
            float4 v = make_float4(0.f, 0.f, 0.f, 0.f);
            if (inb) v = *(const float4*)(x + (long)gr * KGRID + k0);
            *(bfx4*)(&As[buf][row][akg]) = cvt4(v.x, v.y, v.z, v.w);
        }
    };

    stageB(0, 0);
    stageAFull(0, 0);
    wait_async0();
    __syncthreads();

    for (int ks = 0; ks < K1_TILES; ++ks) {
        const int cur = ks & 1;
        const int nb  = cur ^ 1;
        if (ks + 1 < K1_TILES - 1) {           // steady state: branch-free staging
            stageB(ks + 1, nb);
            stageAFull(ks + 1, nb);
        } else if (ks + 1 == K1_TILES - 1) {   // peeled tail staging (once)
            stageB(ks + 1, nb);
            stageATail(nb);
        }

        v16bf af[4];
#pragma unroll
        for (int sm = 0; sm < 4; ++sm)
            af[sm] = load_frag(&As[cur][wm * 64 + sm * 16 + l16][khalf]);
        v16bf bfr[4];
#pragma unroll
        for (int sn = 0; sn < 4; ++sn)
            bfr[sn] = load_frag(&Bs[cur][wn * 64 + sn * 16 + l16][khalf]);

#pragma unroll
        for (int sm = 0; sm < 4; ++sm)
#pragma unroll
            for (int sn = 0; sn < 4; ++sn)
                acc[sm][sn] = __builtin_amdgcn_wmma_f32_16x16x32_bf16(
                    false, af[sm], false, bfr[sn], (short)0, acc[sm][sn], false, false);

        wait_async0();
        __syncthreads();
    }

    // epilogue: + b1g, ELU, bf16 -> t[:, :384]; guard only in the last row-block
    const int mhi = (lane >> 4) * 8;
    auto epi = [&](bool guard) {
#pragma unroll
        for (int sn = 0; sn < 4; ++sn) {
            int col  = colBase + wn * 64 + sn * 16 + l16;
            float bb = b1g[col];
#pragma unroll
            for (int sm = 0; sm < 4; ++sm) {
#pragma unroll
                for (int r = 0; r < 8; ++r) {
                    int row = rowBase + wm * 64 + sm * 16 + mhi + r;
                    if (!guard || row < N_NODES)
                        t[(long)row * K2_DIM + col] = f2bf(elu1(acc[sm][sn][r] + bb));
                }
            }
        }
    };
    if (rowBase + BM <= N_NODES) epi(false); else epi(true);
}

// ---------------------------------------------------------------------------
// GEMM2: out = mask ? t @ [W2g;W2e] + (b2g+b2e) : 0
//   A (t, bf16) and B (W2T, bf16) both staged via async global->LDS copies.
// ---------------------------------------------------------------------------
__global__ __launch_bounds__(NT) void face_gemm2_kernel(
    const __bf16* __restrict__ t, const __bf16* __restrict__ W2T,
    const float* __restrict__ b2g, const float* __restrict__ b2e,
    const float* __restrict__ is_face, float* __restrict__ out)
{
    __shared__ alignas(16) __bf16 As[2][BM][LDST];
    __shared__ alignas(16) __bf16 Bs[2][BN][LDST];

    const int tid  = threadIdx.x;
    const int lane = tid & 31;
    const int wave = tid >> 5;
    const int wm   = wave & 1;
    const int wn   = wave >> 1;
    const int l16  = lane & 15;
    const int khalf = (lane >> 4) * 8;

    const int colBase = blockIdx.x * BN;
    const int rowBase = blockIdx.y * BM;

    v8f acc[4][4] = {};

    auto stage = [&](int ks, int buf) {
#pragma unroll
        for (int p = 0; p < 4; ++p) {
            int idx = tid + NT * p;                    // 0..511
            int r   = idx >> 2;                        // 0..127
            int g   = (idx & 3) * 8;                   // 0,8,16,24
            async_b128(t + (long)(rowBase + r) * K2_DIM + ks * BK + g,
                       &As[buf][r][g]);
            async_b128(W2T + (long)(colBase + r) * K2_DIM + ks * BK + g,
                       &Bs[buf][r][g]);
        }
    };

    stage(0, 0);
    wait_async0();
    __syncthreads();

    for (int ks = 0; ks < K2_TILES; ++ks) {
        const int cur = ks & 1;
        if (ks + 1 < K2_TILES) stage(ks + 1, cur ^ 1);

        v16bf af[4];
#pragma unroll
        for (int sm = 0; sm < 4; ++sm)
            af[sm] = load_frag(&As[cur][wm * 64 + sm * 16 + l16][khalf]);
        v16bf bfr[4];
#pragma unroll
        for (int sn = 0; sn < 4; ++sn)
            bfr[sn] = load_frag(&Bs[cur][wn * 64 + sn * 16 + l16][khalf]);

#pragma unroll
        for (int sm = 0; sm < 4; ++sm)
#pragma unroll
            for (int sn = 0; sn < 4; ++sn)
                acc[sm][sn] = __builtin_amdgcn_wmma_f32_16x16x32_bf16(
                    false, af[sm], false, bfr[sn], (short)0, acc[sm][sn], false, false);

        wait_async0();
        __syncthreads();
    }

    // epilogue: + (b2g+b2e), face mask, f32 store; guard only in last row-block
    const int mhi = (lane >> 4) * 8;
    auto epi = [&](bool guard) {
#pragma unroll
        for (int sn = 0; sn < 4; ++sn) {
            int col  = colBase + wn * 64 + sn * 16 + l16;
            float bb = b2g[col] + b2e[col];
#pragma unroll
            for (int sm = 0; sm < 4; ++sm) {
#pragma unroll
                for (int r = 0; r < 8; ++r) {
                    int row = rowBase + wm * 64 + sm * 16 + mhi + r;
                    if (!guard || row < N_NODES) {
                        float m = is_face[row];
                        out[(long)row * HID + col] = (m > 0.5f) ? (acc[sm][sn][r] + bb) : 0.0f;
                    }
                }
            }
        }
    };
    if (rowBase + BM <= N_NODES) epi(false); else epi(true);
}

// ---------------------------------------------------------------------------
extern "C" void kernel_launch(void* const* d_in, const int* in_sizes, int n_in,
                              void* d_out, int out_size, void* d_ws, size_t ws_size,
                              hipStream_t stream) {
    const float* x[2]    = { (const float*)d_in[0], (const float*)d_in[4] };
    const float* isf[2]  = { (const float*)d_in[1], (const float*)d_in[5] };
    const int*   et[2]   = { (const int*)d_in[2],   (const int*)d_in[6] };
    const float* area[2] = { (const float*)d_in[3], (const float*)d_in[7] };
    const float* W1g = (const float*)d_in[8];
    const float* b1g = (const float*)d_in[9];
    const float* W2g = (const float*)d_in[10];
    const float* b2g = (const float*)d_in[11];
    const float* W1e = (const float*)d_in[12];
    const float* b1e = (const float*)d_in[13];
    const float* W2e = (const float*)d_in[14];
    const float* b2e = (const float*)d_in[15];

    // ws layout: t (NPAD x 768 bf16, ~154 MB) | W1T (384x704 bf16) | W2T (384x768 bf16)
    __bf16* t   = (__bf16*)d_ws;
    __bf16* W1T = t + (long)NPAD * K2_DIM;
    __bf16* W2T = W1T + (long)HID * K1_PAD;
    float* out  = (float*)d_out;

    face_prep_w1_kernel<<<(HID * K1_PAD + 255) / 256, 256, 0, stream>>>(W1g, W1T);
    face_prep_w2_kernel<<<(HID * K2_DIM + 255) / 256, 256, 0, stream>>>(W2g, W2e, W2T);

    dim3 gemmGrid(HID / BN, ROW_BLKS);          // (3 col-blocks, 782 row-blocks)
    int entBlocks = (int)(((long)N_NODES * (HID / 4) + 255) / 256);

    for (int g = 0; g < 2; ++g) {
        face_entity_kernel<<<entBlocks, 256, 0, stream>>>(et[g], area[g], W1e, b1e, t);
        face_gemm1_kernel<<<gemmGrid, NT, 0, stream>>>(x[g], W1T, b1g, t);
        face_gemm2_kernel<<<gemmGrid, NT, 0, stream>>>(t, W2T, b2g, b2e, isf[g],
                                                       out + (long)g * N_NODES * HID);
    }
}
